// MultiHeadAttention_7559142441104
// MI455X (gfx1250) — matrix operand order repaired and checked by reading, not compiled, not run on hardware
//
#include <hip/hip_runtime.h>
#include <hip/hip_bf16.h>
#include <stddef.h>

// ---------------------------------------------------------------------------
// Shapes from the reference
// ---------------------------------------------------------------------------
#define BATCH 16
#define SEQ   600          // 25*24
#define NHEAD 12           // 2*2*3
#define HD    64           // 4*4*4
#define DTOK  768          // 8*8*12
#define NBH   (BATCH*NHEAD)        // 192
#define QTILES32 19                // ceil(600/32): 32-query-row tiles
#define KVTILES  19                // ceil(600/32): 32-key tiles
#define VSTRIDE 608                // padded seq stride for vT (32B-aligned rows)

typedef __attribute__((ext_vector_type(16))) __bf16 v16bf;
typedef __attribute__((ext_vector_type(8)))  __bf16 v8bf;
typedef __attribute__((ext_vector_type(8)))  float  v8f;

#define WMMA_BF16(A, B, C) \
  __builtin_amdgcn_wmma_f32_16x16x32_bf16(false, (A), false, (B), (short)0, (C), false, false)

// ---------------------------------------------------------------------------
// Kernel 1: fused Q/K/V Tucker mode-product projections.
// One block (256 thr) per token. Scatters into head-major bf16 buffers:
//   q2[bh][s][d] (pre-scaled by HD^-0.5), k2[bh][s][d], vT2[bh][d][s_pad608]
// ---------------------------------------------------------------------------
__global__ __launch_bounds__(256) void tle_qkv_kernel(
    const float* __restrict__ x,
    const float* __restrict__ Wq1, const float* __restrict__ Wq2,
    const float* __restrict__ Wq3, const float* __restrict__ bq,
    const float* __restrict__ Wk1, const float* __restrict__ Wk2,
    const float* __restrict__ Wk3, const float* __restrict__ bk,
    const float* __restrict__ Wv1, const float* __restrict__ Wv2,
    const float* __restrict__ Wv3, const float* __restrict__ bv,
    __bf16* __restrict__ q2, __bf16* __restrict__ k2, __bf16* __restrict__ vT2)
{
  __shared__ float xs[DTOK], t1[DTOK], t2[DTOK];
  __shared__ float W1s[64], W2s[64], W3s[144];

  const int token = blockIdx.x;
  const int b = token / SEQ;
  const int s = token % SEQ;
  const int tid = threadIdx.x;

  for (int u = tid; u < DTOK; u += 256) xs[u] = x[(size_t)token * DTOK + u];

  const float* W1p[3] = {Wq1, Wk1, Wv1};
  const float* W2p[3] = {Wq2, Wk2, Wv2};
  const float* W3p[3] = {Wq3, Wk3, Wv3};
  const float* bp [3] = {bq,  bk,  bv };

  for (int p = 0; p < 3; ++p) {
    __syncthreads();                       // protect weight/t1/t2 reuse
    for (int u = tid; u < 64;  u += 256) { W1s[u] = W1p[p][u]; W2s[u] = W2p[p][u]; }
    for (int u = tid; u < 144; u += 256) W3s[u] = W3p[p][u];
    __syncthreads();

    // mode-3: t1[i,j,c] = sum_k xs[i,j,k] * W3[c,k]
    for (int u = tid; u < DTOK; u += 256) {
      const int c = u % 12, ij = u / 12;
      float acc = 0.f;
      #pragma unroll
      for (int k = 0; k < 12; ++k) acc += xs[ij * 12 + k] * W3s[c * 12 + k];
      t1[u] = acc;
    }
    __syncthreads();

    // mode-2: t2[i,b2,c] = sum_j t1[i,j,c] * W2[b2,j]
    for (int u = tid; u < DTOK; u += 256) {
      const int c = u % 12, b2 = (u / 12) % 8, i = u / 96;
      float acc = 0.f;
      #pragma unroll
      for (int j = 0; j < 8; ++j) acc += t1[(i * 8 + j) * 12 + c] * W2s[b2 * 8 + j];
      t2[u] = acc;
    }
    __syncthreads();

    // mode-1 + bias, then split_heads scatter
    for (int u = tid; u < DTOK; u += 256) {
      const int c = u % 12, b2 = (u / 12) % 8, a = u / 96;
      float acc = bp[p][u];
      #pragma unroll
      for (int i = 0; i < 8; ++i) acc += t2[(i * 8 + b2) * 12 + c] * W1s[a * 8 + i];

      const int xm = a >> 1, g = a & 1;
      const int ym = b2 >> 1, h = b2 & 1;
      const int zm = c / 3,  f = c % 3;
      const int head = (g * 2 + h) * 3 + f;
      const int hd   = (xm * 4 + ym) * 4 + zm;
      const int bh   = b * NHEAD + head;

      if (p == 0)      q2[((size_t)bh * SEQ + s) * HD + hd] = (__bf16)(acc * 0.125f);
      else if (p == 1) k2[((size_t)bh * SEQ + s) * HD + hd] = (__bf16)acc;
      else             vT2[((size_t)bh * HD + hd) * VSTRIDE + s] = (__bf16)acc;
    }
  }
}

// ---------------------------------------------------------------------------
// Kernel 2: flash attention, one wave32 per (bh, 32-query-row tile).
// 16 WMMAs per 32-key tile: 8 for Q@K^T (two query sub-tiles x two 16-key
// N-tiles x K=64 in two chunks), 8 for P@V (two query sub-tiles x four
// 16-col N-tiles, K=32 per WMMA). K/V B-operand loads are shared by both
// query sub-tiles, halving L2 traffic vs a 16-row tile.
// Online softmax in f32 using 16-lane butterflies (f32 C-layout rows span a
// 16-lane half). P is transposed C-layout -> A-layout through LDS.
// ---------------------------------------------------------------------------
__global__ __launch_bounds__(32) void flash_attn_kernel(
    const __bf16* __restrict__ q2, const __bf16* __restrict__ k2,
    const __bf16* __restrict__ vT2, float* __restrict__ y2)
{
  __shared__ __attribute__((aligned(32))) __bf16 Pld[32][32];

  const int bh = blockIdx.x / QTILES32;
  const int qt = blockIdx.x % QTILES32;
  const int q0 = qt * 32;
  const int lane  = threadIdx.x;
  const int lhalf = (lane < 16) ? 0 : 1;
  const int lmod  = lane & 15;

  const __bf16* qptr = q2  + (size_t)bh * SEQ * HD;
  const __bf16* kptr = k2  + (size_t)bh * SEQ * HD;
  const __bf16* vptr = vT2 + (size_t)bh * HD * VSTRIDE;

  // ---- Q block (32x64) -> A-layout: [query sub-tile][K-chunk] ----
  v16bf aq[2][2];
  #pragma unroll
  for (int qq = 0; qq < 2; ++qq) {
    int qrow = q0 + qq * 16 + lmod; if (qrow > SEQ - 1) qrow = SEQ - 1;
    const __bf16* qr = qptr + (size_t)qrow * HD;
    const int c0 = lhalf * 16;
    const v8bf lo0 = *(const v8bf*)(qr + c0);
    const v8bf hi0 = *(const v8bf*)(qr + 8 + c0);
    const v8bf lo1 = *(const v8bf*)(qr + 32 + c0);
    const v8bf hi1 = *(const v8bf*)(qr + 40 + c0);
    #pragma unroll
    for (int e = 0; e < 8; ++e) {
      aq[qq][0][e] = lo0[e]; aq[qq][0][e + 8] = hi0[e];
      aq[qq][1][e] = lo1[e]; aq[qq][1][e + 8] = hi1[e];
    }
  }

  v8f O[2][4];
  float m[2][8], l[2][8];
  #pragma unroll
  for (int qq = 0; qq < 2; ++qq) {
    #pragma unroll
    for (int nt = 0; nt < 4; ++nt) O[qq][nt] = (v8f){};
    #pragma unroll
    for (int i = 0; i < 8; ++i) { m[qq][i] = -3.0e38f; l[qq][i] = 0.f; }
  }

  for (int t = 0; t < KVTILES; ++t) {
    const int kbase = t * 32;

    // ---- K B-operands (shared by both query sub-tiles) ----
    int keyA = kbase + lmod;      if (keyA > SEQ - 1) keyA = SEQ - 1;
    int keyB = kbase + 16 + lmod; if (keyB > SEQ - 1) keyB = SEQ - 1;
    const int doff = lhalf * 16;
    const v16bf bA0 = *(const v16bf*)(kptr + (size_t)keyA * HD + doff);
    const v16bf bA1 = *(const v16bf*)(kptr + (size_t)keyA * HD + 32 + doff);
    const v16bf bB0 = *(const v16bf*)(kptr + (size_t)keyB * HD + doff);
    const v16bf bB1 = *(const v16bf*)(kptr + (size_t)keyB * HD + 32 + doff);

    // ---- scores S = Q @ K^T for 32 queries x 32 keys ----
    v8f s[2][2];
    #pragma unroll
    for (int qq = 0; qq < 2; ++qq) {
      v8f s0 = {}, s1 = {};
      s0 = WMMA_BF16(aq[qq][0], bA0, s0);
      s0 = WMMA_BF16(aq[qq][1], bA1, s0);
      s1 = WMMA_BF16(aq[qq][0], bB0, s1);
      s1 = WMMA_BF16(aq[qq][1], bB1, s1);
      s[qq][0] = s0; s[qq][1] = s1;
    }

    // mask out-of-range keys (column index = lane%16 within each N-tile)
    if (kbase + lmod >= SEQ) {
      #pragma unroll
      for (int qq = 0; qq < 2; ++qq)
        #pragma unroll
        for (int i = 0; i < 8; ++i) s[qq][0][i] = -3.0e38f;
    }
    if (kbase + 16 + lmod >= SEQ) {
      #pragma unroll
      for (int qq = 0; qq < 2; ++qq)
        #pragma unroll
        for (int i = 0; i < 8; ++i) s[qq][1][i] = -3.0e38f;
    }

    // ---- online softmax (rows live in 16-lane halves) ----
    float p0[2][8], p1[2][8];
    #pragma unroll
    for (int qq = 0; qq < 2; ++qq) {
      #pragma unroll
      for (int i = 0; i < 8; ++i) {
        float mt = fmaxf(s[qq][0][i], s[qq][1][i]);
        #pragma unroll
        for (int off = 8; off; off >>= 1) mt = fmaxf(mt, __shfl_xor(mt, off, 32));
        const float mnew = fmaxf(m[qq][i], mt);
        const float cf = __expf(m[qq][i] - mnew);
        const float e0 = __expf(s[qq][0][i] - mnew);
        const float e1 = __expf(s[qq][1][i] - mnew);
        float rs = e0 + e1;
        #pragma unroll
        for (int off = 8; off; off >>= 1) rs += __shfl_xor(rs, off, 32);
        l[qq][i] = l[qq][i] * cf + rs;
        m[qq][i] = mnew;
        O[qq][0][i] *= cf; O[qq][1][i] *= cf;
        O[qq][2][i] *= cf; O[qq][3][i] *= cf;
        p0[qq][i] = e0; p1[qq][i] = e1;
      }
    }

    // ---- P: C-layout -> A-layout through LDS (32x32 bf16) ----
    __syncthreads();
    {
      const int rowb = lhalf * 8;
      #pragma unroll
      for (int qq = 0; qq < 2; ++qq) {
        #pragma unroll
        for (int i = 0; i < 8; ++i) {
          Pld[qq * 16 + rowb + i][lmod]      = (__bf16)p0[qq][i];
          Pld[qq * 16 + rowb + i][16 + lmod] = (__bf16)p1[qq][i];
        }
      }
    }
    __syncthreads();
    v16bf pA[2];
    #pragma unroll
    for (int qq = 0; qq < 2; ++qq) {
      const __bf16* pr = &Pld[qq * 16 + lmod][0];
      const int c0 = lhalf * 16;
      const v8bf lo = *(const v8bf*)(pr + c0);
      const v8bf hi = *(const v8bf*)(pr + 8 + c0);
      #pragma unroll
      for (int e = 0; e < 8; ++e) { pA[qq][e] = lo[e]; pA[qq][e + 8] = hi[e]; }
    }

    // ---- O += P @ V  (K=32; V B-operands shared by both query sub-tiles) --
    {
      const __bf16* vrow = vptr + kbase + lhalf * 16;
      const v16bf bv0 = *(const v16bf*)(vrow + (size_t)(lmod)      * VSTRIDE);
      const v16bf bv1 = *(const v16bf*)(vrow + (size_t)(lmod + 16) * VSTRIDE);
      const v16bf bv2 = *(const v16bf*)(vrow + (size_t)(lmod + 32) * VSTRIDE);
      const v16bf bv3 = *(const v16bf*)(vrow + (size_t)(lmod + 48) * VSTRIDE);
      #pragma unroll
      for (int qq = 0; qq < 2; ++qq) {
        O[qq][0] = WMMA_BF16(pA[qq], bv0, O[qq][0]);
        O[qq][1] = WMMA_BF16(pA[qq], bv1, O[qq][1]);
        O[qq][2] = WMMA_BF16(pA[qq], bv2, O[qq][2]);
        O[qq][3] = WMMA_BF16(pA[qq], bv3, O[qq][3]);
      }
    }
  }

  // ---- epilogue: normalize rows and store f32 ----
  #pragma unroll
  for (int qq = 0; qq < 2; ++qq) {
    #pragma unroll
    for (int i = 0; i < 8; ++i) {
      const int row = q0 + qq * 16 + lhalf * 8 + i;
      if (row < SEQ) {
        const float rl = 1.0f / l[qq][i];
        float* yr = y2 + ((size_t)bh * SEQ + row) * HD;
        yr[lmod]      = O[qq][0][i] * rl;
        yr[16 + lmod] = O[qq][1][i] * rl;
        yr[32 + lmod] = O[qq][2][i] * rl;
        yr[48 + lmod] = O[qq][3][i] * rl;
      }
    }
  }
}

// ---------------------------------------------------------------------------
// Kernel 3: merge_heads gather + output TLE + bias -> fp32 output
// ---------------------------------------------------------------------------
__global__ __launch_bounds__(256) void tle_out_kernel(
    const float* __restrict__ y2,
    const float* __restrict__ Wo1, const float* __restrict__ Wo2,
    const float* __restrict__ Wo3, const float* __restrict__ bo,
    float* __restrict__ out)
{
  __shared__ float xs[DTOK], t1[DTOK], t2[DTOK];
  __shared__ float W1s[64], W2s[64], W3s[144];

  const int token = blockIdx.x;
  const int b = token / SEQ;
  const int s = token % SEQ;
  const int tid = threadIdx.x;

  for (int u = tid; u < 64;  u += 256) { W1s[u] = Wo1[u]; W2s[u] = Wo2[u]; }
  for (int u = tid; u < 144; u += 256) W3s[u] = Wo3[u];

  // merge_heads gather into (d1,d2,d3) token layout
  for (int u = tid; u < DTOK; u += 256) {
    const int c = u % 12, b2 = (u / 12) % 8, a = u / 96;
    const int xm = a >> 1, g = a & 1;
    const int ym = b2 >> 1, h = b2 & 1;
    const int zm = c / 3,  f = c % 3;
    const int head = (g * 2 + h) * 3 + f;
    const int hd   = (xm * 4 + ym) * 4 + zm;
    xs[u] = y2[((size_t)(b * NHEAD + head) * SEQ + s) * HD + hd];
  }
  __syncthreads();

  for (int u = tid; u < DTOK; u += 256) {
    const int c = u % 12, ij = u / 12;
    float acc = 0.f;
    #pragma unroll
    for (int k = 0; k < 12; ++k) acc += xs[ij * 12 + k] * W3s[c * 12 + k];
    t1[u] = acc;
  }
  __syncthreads();

  for (int u = tid; u < DTOK; u += 256) {
    const int c = u % 12, b2 = (u / 12) % 8, i = u / 96;
    float acc = 0.f;
    #pragma unroll
    for (int j = 0; j < 8; ++j) acc += t1[(i * 8 + j) * 12 + c] * W2s[b2 * 8 + j];
    t2[u] = acc;
  }
  __syncthreads();

  for (int u = tid; u < DTOK; u += 256) {
    const int c = u % 12, b2 = (u / 12) % 8, a = u / 96;
    float acc = bo[u];
    #pragma unroll
    for (int i = 0; i < 8; ++i) acc += t2[(i * 8 + b2) * 12 + c] * W1s[a * 8 + i];
    out[(size_t)token * DTOK + u] = acc;
  }
}

// ---------------------------------------------------------------------------
// Launcher
// ---------------------------------------------------------------------------
extern "C" void kernel_launch(void* const* d_in, const int* in_sizes, int n_in,
                              void* d_out, int out_size, void* d_ws, size_t ws_size,
                              hipStream_t stream) {
  const float* x   = (const float*)d_in[0];
  const float* Wq1 = (const float*)d_in[1];
  const float* Wq2 = (const float*)d_in[2];
  const float* Wq3 = (const float*)d_in[3];
  const float* bq  = (const float*)d_in[4];
  const float* Wk1 = (const float*)d_in[5];
  const float* Wk2 = (const float*)d_in[6];
  const float* Wk3 = (const float*)d_in[7];
  const float* bk  = (const float*)d_in[8];
  const float* Wv1 = (const float*)d_in[9];
  const float* Wv2 = (const float*)d_in[10];
  const float* Wv3 = (const float*)d_in[11];
  const float* bv  = (const float*)d_in[12];
  const float* Wo1 = (const float*)d_in[13];
  const float* Wo2 = (const float*)d_in[14];
  const float* Wo3 = (const float*)d_in[15];
  const float* bo  = (const float*)d_in[16];
  float* out = (float*)d_out;

  // workspace layout (all 32B-aligned)
  char* ws = (char*)d_ws;
  const size_t QK_ELEMS = (size_t)NBH * SEQ * HD;            // 7,372,800
  const size_t QOFF = 0;
  const size_t KOFF = QOFF + QK_ELEMS * sizeof(__bf16);      // 14,745,600
  const size_t VOFF = KOFF + QK_ELEMS * sizeof(__bf16);      // 29,491,200
  const size_t VSZ  = (size_t)NBH * HD * VSTRIDE * sizeof(__bf16);
  const size_t YOFF = VOFF + VSZ;                            // 44,433,408
  __bf16* q2  = (__bf16*)(ws + QOFF);
  __bf16* k2  = (__bf16*)(ws + KOFF);
  __bf16* vT2 = (__bf16*)(ws + VOFF);
  float*  y2  = (float*)(ws + YOFF);

  tle_qkv_kernel<<<BATCH * SEQ, 256, 0, stream>>>(
      x, Wq1, Wq2, Wq3, bq, Wk1, Wk2, Wk3, bk, Wv1, Wv2, Wv3, bv, q2, k2, vT2);

  flash_attn_kernel<<<NBH * QTILES32, 32, 0, stream>>>(q2, k2, vT2, y2);

  tle_out_kernel<<<BATCH * SEQ, 256, 0, stream>>>(y2, Wo1, Wo2, Wo3, bo, out);
}
